// EmaHead_24635932410001
// MI455X (gfx1250) — matrix operand order repaired and checked by reading, not compiled, not run on hardware
//
#include <hip/hip_runtime.h>
#include <math.h>

// Problem constants (from reference)
#define T_DIM 4096
#define B_DIM 256
#define FEAT  128
#define HID   16
#define M_ROWS (T_DIM * B_DIM)      // 1,048,576 rows of the MLP "GEMM"
#define TILES  (M_ROWS / 16)        // 65,536 16-row WMMA tiles
#define G_CHUNKS    64              // chunks for parallel scan over T
#define T_PER_CHUNK (T_DIM / G_CHUNKS) // 64

typedef __attribute__((ext_vector_type(16))) __bf16 bf16x16;
typedef __attribute__((ext_vector_type(8)))  float  f32x8;

// ---------------------------------------------------------------------------
// Kernel 1: alpha = clamp(sigmoid(relu(feat @ W1 + b1) @ W2 + b2), .01, .99)
// One wave computes one 16x16 (rows x HID) tile via 4x v_wmma_f32_16x16x32_bf16.
// A layout (16-bit A 16x32): lanes 0-15 & 16-31 both hold M = lane&15;
//   half = lane>>4 selects K sub-block: elems [0..7] = K kb..kb+7,
//   elems [8..15] = K kb+16..kb+23, kb = chunk*32 + half*8.
// B layout (32x16): lane holds column N = lane&15, same K order as A.
// C layout: VGPR j, lanes 0-15 -> row j, lanes 16-31 -> row j+8; N = lane&15.
// ---------------------------------------------------------------------------
__global__ void __launch_bounds__(256) alpha_mlp_wmma(
    const float* __restrict__ feat, const float* __restrict__ W1,
    const float* __restrict__ b1,   const float* __restrict__ W2,
    const float* __restrict__ b2,   float* __restrict__ alpha) {
  const int lane         = threadIdx.x & 31;
  const int waveInBlock  = threadIdx.x >> 5;
  const int wavesPerBlk  = blockDim.x >> 5;
  const int waveId       = blockIdx.x * wavesPerBlk + waveInBlock;
  const int nWaves       = gridDim.x * wavesPerBlk;

  const int half = lane >> 4;   // K sub-block selector
  const int m    = lane & 15;   // A-row-in-tile / B,C column

  // --- B fragments for W1 [K=128][N=16] row-major: load once, reuse ---
  bf16x16 bfrag[4];
  #pragma unroll
  for (int c = 0; c < 4; ++c) {
    const int kbase = c * 32 + half * 8;
    #pragma unroll
    for (int e = 0; e < 16; ++e)
      bfrag[c][e] = (__bf16)W1[(kbase + (e < 8 ? e : e + 8)) * HID + m];
  }
  const float bias1 = b1[m];
  const float w2    = W2[m];
  const float bias2 = b2[0];

  for (int tile = waveId; tile < TILES; tile += nWaves) {
    const int   row0 = tile * 16;
    const float* rowp = feat + (size_t)(row0 + m) * FEAT;

    // Prefetch this wave's next tile (emits global_prefetch_b8)
    __builtin_prefetch(rowp + (size_t)nWaves * 16 * FEAT, 0, 1);

    f32x8 acc = {};
    #pragma unroll
    for (int c = 0; c < 4; ++c) {
      const float* p = rowp + c * 32 + half * 8;
      float4 f0 = ((const float4*)p)[0];
      float4 f1 = ((const float4*)p)[1];
      float4 f2 = ((const float4*)(p + 16))[0];
      float4 f3 = ((const float4*)(p + 16))[1];
      bf16x16 a;
      a[0]  = (__bf16)f0.x; a[1]  = (__bf16)f0.y; a[2]  = (__bf16)f0.z; a[3]  = (__bf16)f0.w;
      a[4]  = (__bf16)f1.x; a[5]  = (__bf16)f1.y; a[6]  = (__bf16)f1.z; a[7]  = (__bf16)f1.w;
      a[8]  = (__bf16)f2.x; a[9]  = (__bf16)f2.y; a[10] = (__bf16)f2.z; a[11] = (__bf16)f2.w;
      a[12] = (__bf16)f3.x; a[13] = (__bf16)f3.y; a[14] = (__bf16)f3.z; a[15] = (__bf16)f3.w;
      acc = __builtin_amdgcn_wmma_f32_16x16x32_bf16(
          /*neg_a=*/false, a, /*neg_b=*/false, bfrag[c],
          /*c_mod=*/(short)0, acc, /*reuse_a=*/false, /*reuse_b=*/false);
    }

    // Epilogue: bias + ReLU + per-lane product with W2, then a butterfly
    // reduction over the 16 N-lanes. Stage-major order so the 8 ds_bpermute
    // of one stage clause together with a single s_wait_dscnt (vs. 32
    // serialized round-trips in the row-major order).
    float p[8];
    #pragma unroll
    for (int j = 0; j < 8; ++j) {
      float h = fmaxf(acc[j] + bias1, 0.0f);
      p[j] = h * w2;
    }
    #pragma unroll
    for (int s = 1; s < 16; s <<= 1) {
      float t[8];
      #pragma unroll
      for (int j = 0; j < 8; ++j) t[j] = __shfl_xor(p[j], s, 16);
      #pragma unroll
      for (int j = 0; j < 8; ++j) p[j] += t[j];
    }

    // Lane 0 now holds row sums for rows row0+0..7 (p[0..7]); lane 16 holds
    // rows row0+8..15. Fast sigmoid (v_rcp_f32; alpha is clamped anyway),
    // then two contiguous float4 stores per tile.
    if (m == 0) {
      float s[8];
      #pragma unroll
      for (int j = 0; j < 8; ++j) {
        float v = __builtin_amdgcn_rcpf(1.0f + __expf(-(p[j] + bias2)));
        s[j] = fminf(fmaxf(v, 0.01f), 0.99f);
      }
      float4* outp = (float4*)(alpha + row0 + half * 8);
      outp[0] = make_float4(s[0], s[1], s[2], s[3]);
      outp[1] = make_float4(s[4], s[5], s[6], s[7]);
    }
  }
}

// ---------------------------------------------------------------------------
// Parallel scan of y[t] = (1-a)y[t-1] + a r[t]  (affine composition).
// Pass 1: per-chunk composite (A, B) per batch lane.
// ---------------------------------------------------------------------------
__global__ void __launch_bounds__(256) ema_chunk_reduce(
    const float* __restrict__ alpha, const float* __restrict__ r,
    float2* __restrict__ chunkAB) {
  const int b = threadIdx.x;          // batch lane (coalesced across threads)
  const int g = blockIdx.x;           // chunk
  float A = 0.0f, Bc = 1.0f;
  const int t0 = g * T_PER_CHUNK;
  for (int i = 0; i < T_PER_CHUNK; ++i) {
    const int t = t0 + i;
    const float a  = alpha[(size_t)t * B_DIM + b];
    const float rt = r[(size_t)t * B_DIM + b];
    if (t == 0) { A = rt; Bc = 0.0f; }             // y[0] = r[0] (resets state)
    else        { A = (1.0f - a) * A + a * rt; Bc *= (1.0f - a); }
  }
  chunkAB[g * B_DIM + b] = make_float2(A, Bc);
}

// Pass 2: exclusive scan of the 64 chunk composites -> incoming y per chunk.
__global__ void __launch_bounds__(256) ema_chunk_scan(
    const float2* __restrict__ chunkAB, float* __restrict__ yin) {
  const int b = threadIdx.x;
  float y = 0.0f;
  yin[b] = 0.0f;                       // chunk 0: t==0 resets, value unused
  for (int g = 1; g < G_CHUNKS; ++g) {
    float2 c = chunkAB[(g - 1) * B_DIM + b];
    y = c.x + c.y * y;
    yin[g * B_DIM + b] = y;
  }
}

// Pass 3: replay each chunk from its incoming state and emit outputs.
__global__ void __launch_bounds__(256) ema_chunk_apply(
    const float* __restrict__ alpha, const float* __restrict__ r,
    const float* __restrict__ yin,   float* __restrict__ out) {
  const int b = threadIdx.x;
  const int g = blockIdx.x;
  float y = yin[g * B_DIM + b];
  const int t0 = g * T_PER_CHUNK;
  for (int i = 0; i < T_PER_CHUNK; ++i) {
    const int t = t0 + i;
    const float a  = alpha[(size_t)t * B_DIM + b];
    const float rt = r[(size_t)t * B_DIM + b];
    if (t == 0) y = rt;
    else        y = (1.0f - a) * y + a * rt;
    out[(size_t)t * B_DIM + b] = y;
  }
}

// ---------------------------------------------------------------------------
extern "C" void kernel_launch(void* const* d_in, const int* in_sizes, int n_in,
                              void* d_out, int out_size, void* d_ws, size_t ws_size,
                              hipStream_t stream) {
  // setup_inputs() order: r, feat, W1, b1, W2, b2 (all float32)
  const float* r    = (const float*)d_in[0];
  const float* feat = (const float*)d_in[1];
  const float* W1   = (const float*)d_in[2];
  const float* b1   = (const float*)d_in[3];
  const float* W2   = (const float*)d_in[4];
  const float* b2   = (const float*)d_in[5];
  float* out = (float*)d_out;

  // Workspace layout (all fully written before read; no cross-call state):
  //   alpha   : M_ROWS floats            (4 MiB)
  //   chunkAB : G_CHUNKS*B_DIM float2    (128 KiB)
  //   yin     : G_CHUNKS*B_DIM floats    (64 KiB)
  float*  alpha   = (float*)d_ws;
  float2* chunkAB = (float2*)(alpha + (size_t)M_ROWS);
  float*  yin     = (float*)(chunkAB + (size_t)G_CHUNKS * B_DIM);

  // 2048 blocks x 8 waves = 16384 waves; 4 tiles per wave (grid-stride).
  alpha_mlp_wmma<<<2048, 256, 0, stream>>>(feat, W1, b1, W2, b2, alpha);
  ema_chunk_reduce<<<G_CHUNKS, B_DIM, 0, stream>>>(alpha, r, chunkAB);
  ema_chunk_scan<<<1, B_DIM, 0, stream>>>(chunkAB, yin);
  ema_chunk_apply<<<G_CHUNKS, B_DIM, 0, stream>>>(alpha, r, yin, out);
}
